// SelfAttention_32555852104190
// MI455X (gfx1250) — matrix operand order repaired and checked by reading, not compiled, not run on hardware
//
#include <hip/hip_runtime.h>

typedef __attribute__((ext_vector_type(16))) __bf16 v16bf;
typedef __attribute__((ext_vector_type(8)))  __bf16 bf16x8;
typedef __attribute__((ext_vector_type(8)))  float  v8f;

#define S_LEN 1024   // H*W
#define C_CH  256
#define HD    64
#define NH    4
#define BATCH 16

// ---- WMMA fragment loaders (CDNA5 16x16x32 bf16 layouts, CDNA5 ISA 7.12.2) ----
// A (16x32, MxK): lane m=l&15, half=l>>4; elems 0..7 -> K = half*8+0..7, elems 8..15 -> K = 16+half*8+0..7
__device__ __forceinline__ v16bf load_a_bf16(const __bf16* base, int ld, int lane) {
    int m = lane & 15, half = lane >> 4;
    const __bf16* p = base + m * ld + half * 8;
    bf16x8 lo = *(const bf16x8*)(p);
    bf16x8 hi = *(const bf16x8*)(p + 16);
    return __builtin_shufflevector(lo, hi, 0,1,2,3,4,5,6,7,8,9,10,11,12,13,14,15);
}

// B (32x16, KxN), memory n-major: element (K,n) at base[n*ld + K]; lane n=l&15, half=l>>4; elems 0..7 -> K = half*8+0..7, elems 8..15 -> K = 16+half*8+0..7
__device__ __forceinline__ v16bf load_b_bf16(const __bf16* base, int ld, int lane) {
    int n = lane & 15, half = lane >> 4;
    const __bf16* p = base + n * ld + half * 8;
    bf16x8 lo = *(const bf16x8*)(p);
    bf16x8 hi = *(const bf16x8*)(p + 16);
    return __builtin_shufflevector(lo, hi, 0,1,2,3,4,5,6,7,8,9,10,11,12,13,14,15);
}

__device__ __forceinline__ v8f wmma_bf16(v16bf a, v16bf b, v8f c) {
    return __builtin_amdgcn_wmma_f32_16x16x32_bf16(false, a, false, b, (short)0, c, false, false);
}

// ---- Kernel P: one-shot fp32 -> bf16 weight conversion (removes cvt from GEMM hot loops) ----
__global__ void wcvt_kernel(const float* __restrict__ qkv_w,
                            const float* __restrict__ out_w,
                            __bf16* __restrict__ wq,    // 768*256
                            __bf16* __restrict__ wo) {  // 256*256
    int gid = blockIdx.x * blockDim.x + threadIdx.x;    // 262144 threads
    if (gid < 768 * C_CH) wq[gid] = (__bf16)qkv_w[gid];
    else                  wo[gid - 768 * C_CH] = (__bf16)out_w[gid - 768 * C_CH];
}

// ---- Kernel 0: inference BN, transpose to (B, S, C) bf16 ----
__global__ void bn_transpose_kernel(const float* __restrict__ x,
                                    const float* __restrict__ gamma,
                                    const float* __restrict__ beta,
                                    const float* __restrict__ rmean,
                                    const float* __restrict__ rvar,
                                    __bf16* __restrict__ xt) {
    int gid = blockIdx.x * blockDim.x + threadIdx.x;   // B*C*S threads
    int s = gid & (S_LEN - 1);
    int c = (gid >> 10) & (C_CH - 1);
    int b = gid >> 18;
    float inv = rsqrtf(rvar[c] + 1e-5f);
    float sc = inv * gamma[c];
    float sh = beta[c] - rmean[c] * sc;
    float v = x[gid] * sc + sh;
    xt[((size_t)b * S_LEN + s) * C_CH + c] = (__bf16)v;
}

// ---- Kernel 1: QKV GEMM (768 x 256) @ (256 x 1024) per batch ----
// Each wave: 16(M) x 64(N) output block; A fragment reused across 4 B tiles.
__global__ __launch_bounds__(256) void qkv_gemm_kernel(
        const __bf16* __restrict__ w,    // 768x256 bf16 row-major
        const float* __restrict__ bias,  // 768
        const __bf16* __restrict__ xt,   // (B,S,C)
        __bf16* __restrict__ qbuf,       // (B*NH, S, HD)
        __bf16* __restrict__ kbuf,       // (B*NH, S, HD)
        __bf16* __restrict__ vbuf) {     // (B*NH, HD, S)
    int lane = threadIdx.x & 31;
    int wgid = blockIdx.x * (blockDim.x >> 5) + (threadIdx.x >> 5);
    int b  = wgid / 768;                 // 48 M-tiles * 16 N-groups per batch
    int r  = wgid % 768;
    int o0 = (r >> 4) << 4;              // 0..752, step 16
    int s0 = (r & 15) << 6;              // 0..960, step 64
    const __bf16* abase = w + o0 * C_CH;
    const __bf16* bbase = xt + ((size_t)b * S_LEN + s0) * C_CH;
    v8f acc[4] = {{}, {}, {}, {}};
    #pragma unroll
    for (int c0 = 0; c0 < C_CH; c0 += 32) {
        v16bf a = load_a_bf16(abase + c0, C_CH, lane);
        #pragma unroll
        for (int j = 0; j < 4; ++j) {
            v16bf bb = load_b_bf16(bbase + (size_t)(j << 4) * C_CH + c0, C_CH, lane);
            acc[j] = wmma_bf16(a, bb, acc[j]);
        }
    }
    int half = lane >> 4, n = lane & 15;
    #pragma unroll
    for (int j = 0; j < 4; ++j) {
        int s = s0 + (j << 4) + n;
        #pragma unroll
        for (int rr = 0; rr < 8; ++rr) {
            int o = o0 + rr + 8 * half;
            float val = acc[j][rr] + bias[o];
            int which = o >> 8;
            int cf = o & 255;
            int head = cf >> 6;
            int cl = cf & 63;
            int bh = (b << 2) + head;
            __bf16 bv = (__bf16)val;
            if (which == 0)      qbuf[((size_t)bh * S_LEN + s) * HD + cl] = bv;
            else if (which == 1) kbuf[((size_t)bh * S_LEN + s) * HD + cl] = bv;
            else                 vbuf[((size_t)bh * HD + cl) * S_LEN + s] = bv;
        }
    }
}

// ---- Kernel 2: attention per (b, head, 16-query block) ----
__global__ __launch_bounds__(256) void attn_kernel(
        const __bf16* __restrict__ qbuf,
        const __bf16* __restrict__ kbuf,
        const __bf16* __restrict__ vbuf,
        __bf16* __restrict__ av) {       // (B, S, C)
    __shared__ __align__(32) __bf16 s_probs[16 * S_LEN]; // 32 KB
    __shared__ float s_part[8][256];                      // 8 KB
    __shared__ float s_sum[16];

    int tid = threadIdx.x;
    int lane = tid & 31;
    int wid  = tid >> 5;
    int q0 = blockIdx.x << 4;
    int h  = blockIdx.y;
    int b  = blockIdx.z;
    int bh = (b << 2) + h;
    const __bf16* qb = qbuf + ((size_t)bh * S_LEN + q0) * HD;
    const __bf16* kb = kbuf + (size_t)bh * S_LEN * HD;
    const __bf16* vb = vbuf + (size_t)bh * HD * S_LEN;

    // Q block fragments (16 queries x 64 channels = 2 K-fragments), held in registers
    v16bf aq0 = load_a_bf16(qb,      HD, lane);
    v16bf aq1 = load_a_bf16(qb + 32, HD, lane);

    int half = lane >> 4, nn = lane & 15;

    // scores: 8 waves over 64 key blocks
    for (int j = wid; j < 64; j += 8) {
        v8f acc = {};
        v16bf b0 = load_b_bf16(kb + (size_t)(j << 4) * HD,      HD, lane);
        v16bf b1 = load_b_bf16(kb + (size_t)(j << 4) * HD + 32, HD, lane);
        acc = wmma_bf16(aq0, b0, acc);
        acc = wmma_bf16(aq1, b1, acc);
        #pragma unroll
        for (int rr = 0; rr < 8; ++rr) {
            int m = rr + 8 * half;
            s_probs[m * S_LEN + (j << 4) + nn] = (__bf16)(acc[rr] * 0.0625f); // 1/sqrt(256)
        }
    }
    __syncthreads();

    // softmax: 16 lanes per row (contiguous within a wave); unnormalized exp kept
    int row = tid >> 4;
    int t   = tid & 15;
    {
        float mx = -3.0e38f;
        #pragma unroll 4
        for (int i = 0; i < 64; ++i)
            mx = fmaxf(mx, (float)s_probs[row * S_LEN + t + (i << 4)]);
        #pragma unroll
        for (int off = 8; off >= 1; off >>= 1)
            mx = fmaxf(mx, __shfl_xor(mx, off, 16));
        float sum = 0.f;
        #pragma unroll 4
        for (int i = 0; i < 64; ++i) {
            int idx = row * S_LEN + t + (i << 4);
            float e = __expf((float)s_probs[idx] - mx);
            sum += e;
            s_probs[idx] = (__bf16)e;
        }
        #pragma unroll
        for (int off = 8; off >= 1; off >>= 1)
            sum += __shfl_xor(sum, off, 16);
        if (t == 0) s_sum[row] = sum;
    }
    __syncthreads();

    // PV: 8 waves = 4 column tiles x 2 K-halves (K = 1024)
    int col   = wid & 3;
    int khalf = wid >> 2;
    v8f acc = {};
    #pragma unroll
    for (int tt = 0; tt < 16; ++tt) {
        int kk = (khalf << 9) + (tt << 5);
        v16bf a  = load_a_bf16(s_probs + kk, S_LEN, lane);
        v16bf bb = load_b_bf16(vb + (size_t)(col << 4) * S_LEN + kk, S_LEN, lane);
        acc = wmma_bf16(a, bb, acc);
    }
    #pragma unroll
    for (int rr = 0; rr < 8; ++rr) {
        int m = rr + 8 * half;
        s_part[wid][m * 16 + nn] = acc[rr];
    }
    __syncthreads();

    // combine K-halves, normalize, write av (B,S,C) bf16
    for (int idx = tid; idx < 1024; idx += 256) {
        int tile = idx >> 8;
        int e = idx & 255;
        int m = e >> 4;
        int n2 = e & 15;
        float v = (s_part[tile][e] + s_part[tile + 4][e]) / s_sum[m];
        av[((size_t)b * S_LEN + q0 + m) * C_CH + (h << 6) + (tile << 4) + n2] = (__bf16)v;
    }
}

// ---- Kernel 3: output projection + bias + residual ----
// Each wave: 16(M) x 64(N) output block; A fragment reused across 4 B tiles.
__global__ __launch_bounds__(256) void out_proj_kernel(
        const __bf16* __restrict__ w,    // 256x256 bf16
        const float* __restrict__ bias,  // 256
        const __bf16* __restrict__ av,   // (B,S,C)
        const float* __restrict__ x,     // (B,C,S)
        float* __restrict__ out) {
    int lane = threadIdx.x & 31;
    int wgid = blockIdx.x * (blockDim.x >> 5) + (threadIdx.x >> 5);
    int b  = wgid >> 8;                  // 16 M-tiles * 16 N-groups per batch
    int r  = wgid & 255;
    int o0 = (r >> 4) << 4;
    int s0 = (r & 15) << 6;
    const __bf16* abase = w + o0 * C_CH;
    const __bf16* bbase = av + ((size_t)b * S_LEN + s0) * C_CH;
    v8f acc[4] = {{}, {}, {}, {}};
    #pragma unroll
    for (int c0 = 0; c0 < C_CH; c0 += 32) {
        v16bf a = load_a_bf16(abase + c0, C_CH, lane);
        #pragma unroll
        for (int j = 0; j < 4; ++j) {
            v16bf bb = load_b_bf16(bbase + (size_t)(j << 4) * C_CH + c0, C_CH, lane);
            acc[j] = wmma_bf16(a, bb, acc[j]);
        }
    }
    int half = lane >> 4, n = lane & 15;
    #pragma unroll
    for (int j = 0; j < 4; ++j) {
        int s = s0 + (j << 4) + n;
        #pragma unroll
        for (int rr = 0; rr < 8; ++rr) {
            int o = o0 + rr + 8 * half;
            size_t idx = ((size_t)b * C_CH + o) * S_LEN + s;
            out[idx] = x[idx] + bias[o] + acc[j][rr];
        }
    }
}

extern "C" void kernel_launch(void* const* d_in, const int* in_sizes, int n_in,
                              void* d_out, int out_size, void* d_ws, size_t ws_size,
                              hipStream_t stream) {
    (void)in_sizes; (void)n_in; (void)out_size; (void)ws_size;
    const float* x      = (const float*)d_in[0];
    const float* gamma  = (const float*)d_in[1];
    const float* beta   = (const float*)d_in[2];
    const float* rmean  = (const float*)d_in[3];
    const float* rvar   = (const float*)d_in[4];
    const float* qkv_w  = (const float*)d_in[5];
    const float* qkv_b  = (const float*)d_in[6];
    const float* out_w  = (const float*)d_in[7];
    const float* out_b  = (const float*)d_in[8];

    char* ws = (char*)d_ws;
    __bf16* xt   = (__bf16*)(ws);                               // 8 MB, reused as av
    __bf16* qbuf = (__bf16*)(ws + (size_t)(8u  << 20));         // 8 MB
    __bf16* kbuf = (__bf16*)(ws + (size_t)(16u << 20));         // 8 MB
    __bf16* vbuf = (__bf16*)(ws + (size_t)(24u << 20));         // 8 MB
    __bf16* wq   = (__bf16*)(ws + (size_t)(32u << 20));         // 384 KB
    __bf16* wo   = (__bf16*)(ws + (size_t)(32u << 20) + (512u << 10)); // 128 KB

    wcvt_kernel<<<1024, 256, 0, stream>>>(qkv_w, out_w, wq, wo);
    bn_transpose_kernel<<<16384, 256, 0, stream>>>(x, gamma, beta, rmean, rvar, xt);
    qkv_gemm_kernel<<<1536, 256, 0, stream>>>(wq, qkv_b, xt, qbuf, kbuf, vbuf);
    attn_kernel<<<dim3(64, 4, 16), 256, 0, stream>>>(qbuf, kbuf, vbuf, xt /* av, reuses xt */);
    out_proj_kernel<<<512, 256, 0, stream>>>(wo, out_b, xt, x, (float*)d_out);
}
